// Conv_SelfAttn_20074677142273
// MI455X (gfx1250) — compile-verified, matrix operand order checked by reading, not run on hardware
//
#include <hip/hip_runtime.h>

typedef __attribute__((ext_vector_type(16))) _Float16 v16h;
typedef __attribute__((ext_vector_type(8)))  _Float16 v8h;
typedef __attribute__((ext_vector_type(8)))  float    v8f;

#define CB   8      // batches
#define CC   128    // channels (c_in)
#define CQK  16     // q/k channels
#define NN   4096   // spatial positions (64*64)

// ---------------------------------------------------------------------------
// Phase 1a: Q/K projections.  Qh,Kh stored (b, n, 16) f16, row-contiguous so a
// lane can fetch its 8-halve WMMA operand slice with one 16-byte load.
// ---------------------------------------------------------------------------
__global__ void __launch_bounds__(256) qk_proj(
    const float* __restrict__ x, const float* __restrict__ Wq,
    const float* __restrict__ Wk, _Float16* __restrict__ Qh,
    _Float16* __restrict__ Kh) {
  __shared__ float sWq[CQK * CC];
  __shared__ float sWk[CQK * CC];
  int tid = threadIdx.x;
  for (int i = tid; i < CQK * CC; i += 256) { sWq[i] = Wq[i]; sWk[i] = Wk[i]; }
  __syncthreads();

  int gb = blockIdx.x * 256 + tid;          // 0 .. B*N-1
  int b  = gb / NN;
  int n  = gb % NN;
  const float* xp = x + (size_t)b * CC * NN + n;

  float qa[CQK] = {}, ka[CQK] = {};
  for (int c = 0; c < CC; ++c) {
    float xv = xp[(size_t)c * NN];          // coalesced along n
#pragma unroll
    for (int o = 0; o < CQK; ++o) {
      qa[o] += sWq[o * CC + c] * xv;
      ka[o] += sWk[o * CC + c] * xv;
    }
  }
  _Float16* qd = Qh + (size_t)gb * CQK;
  _Float16* kd = Kh + (size_t)gb * CQK;
#pragma unroll
  for (int o = 0; o < CQK; ++o) {
    qd[o] = (_Float16)qa[o];
    kd[o] = (_Float16)ka[o];
  }
}

// ---------------------------------------------------------------------------
// Phase 1b: V projection, stored TRANSPOSED: Vt (b, c, n) f16.
// x-tile (128 x 64, 32 KB) + full Wv (64 KB) staged in LDS so x is read from
// HBM exactly once.  256 threads = 4 channel-groups x 64 columns; weight reads
// are LDS broadcasts within a wave, x reads are conflict-free (stride-1 in n).
// ---------------------------------------------------------------------------
__global__ void __launch_bounds__(256) vt_proj(
    const float* __restrict__ x, const float* __restrict__ Wv,
    _Float16* __restrict__ Vt) {
  __shared__ float xs[CC * 64];
  __shared__ float sW[CC * CC];
  int tid = threadIdx.x;
  int blk = blockIdx.x;                     // 0 .. B*N/64-1
  int b   = blk / (NN / 64);
  int n0  = (blk % (NN / 64)) * 64;
  int nl  = tid & 63;                       // column within tile
  int c0  = (tid >> 6) * 32;                // 32 output channels per thread

  for (int i = tid; i < CC * 64; i += 256) {
    int ci = i >> 6, n = i & 63;
    xs[i] = x[(size_t)b * CC * NN + (size_t)ci * NN + n0 + n];
  }
  for (int i = tid; i < CC * CC; i += 256) sW[i] = Wv[i];
  __syncthreads();

  float acc[32] = {};
  for (int ci = 0; ci < CC; ++ci) {
    float xv = xs[ci * 64 + nl];
#pragma unroll
    for (int j = 0; j < 32; ++j) acc[j] += sW[(c0 + j) * CC + ci] * xv;
  }
#pragma unroll
  for (int j = 0; j < 32; ++j)
    Vt[((size_t)b * CC + (c0 + j)) * NN + n0 + nl] = (_Float16)acc[j];
}

// ---------------------------------------------------------------------------
// Phase 2: flash attention.  One wave per 16-row q-tile; Sᵀ via WMMA so the
// softmax row = lane%16 (one shfl_xor(16) per reduction), Pᵀ packs
// lane-locally, accumulator is outᵀ (c, n) tiles.  Lazy flash rescale: the
// 64-mul accumulator rescale only runs when some lane's running max grew.
// ---------------------------------------------------------------------------
__global__ void __launch_bounds__(256) attn_kernel(
    const float* __restrict__ x, const _Float16* __restrict__ Qh,
    const _Float16* __restrict__ Kh, const _Float16* __restrict__ Vt,
    const float* __restrict__ gamma, float* __restrict__ out) {
  int tid  = threadIdx.x;
  int lane = tid & 31;
  int wave = tid >> 5;
  int qt   = blockIdx.x * 8 + wave;         // global q-tile id (0..2047)
  int b    = qt >> 8;                       // 256 tiles per batch
  int n0   = (qt & 255) << 4;               // q rows n0 .. n0+15
  int r    = lane & 15;                     // q_row within tile / N index
  int g    = lane >> 4;                     // half-group (K-offset selector)

  const v8f zero = {};

  // B-operand: Q tile, B[K=o (padded to 32), N=q_row]; halves 8..15 stay 0.
  v16h bq = {};
  {
    v8h lo = *(const v8h*)(Qh + ((size_t)b * NN + n0 + r) * CQK + 8 * g);
#pragma unroll
    for (int h = 0; h < 8; ++h) bq[h] = lo[h];
  }

  v8f acc[8];
#pragma unroll
  for (int t = 0; t < 8; ++t) acc[t] = zero;
  float mstat = -1e30f, lstat = 0.0f;

  const _Float16* Kb = Kh + (size_t)b * NN * CQK;
  const _Float16* Vb = Vt + (size_t)b * CC * NN;

#pragma unroll 2
  for (int m0 = 0; m0 < NN; m0 += 32) {
    // A-operands: two K tiles, A[M = m col, K = o padded to 32]
    v16h ak0 = {}, ak1 = {};
    {
      v8h k0 = *(const v8h*)(Kb + (size_t)(m0 + r) * CQK + 8 * g);
      v8h k1 = *(const v8h*)(Kb + (size_t)(m0 + 16 + r) * CQK + 8 * g);
#pragma unroll
      for (int h = 0; h < 8; ++h) { ak0[h] = k0[h]; ak1[h] = k1[h]; }
    }
    // Sᵀ tiles: D[M = m local, N = q_row]
    v8f s0 = __builtin_amdgcn_wmma_f32_16x16x32_f16(false, ak0, false, bq,
                                                    (short)0, zero, false, false);
    v8f s1 = __builtin_amdgcn_wmma_f32_16x16x32_f16(false, ak1, false, bq,
                                                    (short)0, zero, false, false);

    // Running max over these 32 columns; per-lane row = r.
    float t = -1e30f;
#pragma unroll
    for (int v = 0; v < 8; ++v) t = fmaxf(t, fmaxf(s0[v], s1[v]));
    t = fmaxf(t, __shfl_xor(t, 16, 32));
    float mnew = fmaxf(mstat, t);

    // Lazy rescale: only when some lane's running max actually grew.
    if (__any(mnew > mstat)) {
      float sc = __expf(mstat - mnew);      // lanes with unchanged max: sc = 1
      lstat *= sc;
#pragma unroll
      for (int ct = 0; ct < 8; ++ct) {
#pragma unroll
        for (int v = 0; v < 8; ++v) acc[ct][v] *= sc;
      }
    }
    mstat = mnew;

    float p0[8], p1[8];
    float rsum = 0.0f;
#pragma unroll
    for (int v = 0; v < 8; ++v) {
      p0[v] = __expf(s0[v] - mnew);
      p1[v] = __expf(s1[v] - mnew);
      rsum += p0[v] + p1[v];
    }
    rsum += __shfl_xor(rsum, 16, 32);
    lstat += rsum;

    // Pack Pᵀ lane-locally into B layout: B[K = m local 0..31, N = q_row].
    v16h bp;
#pragma unroll
    for (int v = 0; v < 8; ++v) {
      bp[v]     = (_Float16)p0[v];
      bp[8 + v] = (_Float16)p1[v];
    }

    // acc += Vᵀ · Pᵀ   (outᵀ tiles, M = c, N = q_row)
#pragma unroll
    for (int ct = 0; ct < 8; ++ct) {
      const _Float16* vrow = Vb + (size_t)(ct * 16 + r) * NN + m0;
      v8h lo = *(const v8h*)(vrow + 8 * g);
      v8h hi = *(const v8h*)(vrow + 16 + 8 * g);
      v16h av;
#pragma unroll
      for (int h = 0; h < 8; ++h) { av[h] = lo[h]; av[8 + h] = hi[h]; }
      acc[ct] = __builtin_amdgcn_wmma_f32_16x16x32_f16(false, av, false, bp,
                                                       (short)0, acc[ct], false, false);
    }
  }

  // Epilogue: out(b,c,n) = x + gamma * outᵀ / lstat  (lstat uniform over v).
  float inv = gamma[0] / lstat;
  const float* xb = x + (size_t)b * CC * NN;
  float* ob = out + (size_t)b * CC * NN;
#pragma unroll
  for (int ct = 0; ct < 8; ++ct) {
#pragma unroll
    for (int v = 0; v < 8; ++v) {
      int c = ct * 16 + v + 8 * g;          // C/D layout: M = v + 8*(lane/16)
      size_t idx = (size_t)c * NN + n0 + r;
      ob[idx] = xb[idx] + inv * acc[ct][v];
    }
  }
}

// ---------------------------------------------------------------------------
extern "C" void kernel_launch(void* const* d_in, const int* in_sizes, int n_in,
                              void* d_out, int out_size, void* d_ws, size_t ws_size,
                              hipStream_t stream) {
  const float* x     = (const float*)d_in[0];
  const float* Wq    = (const float*)d_in[1];
  const float* Wk    = (const float*)d_in[2];
  const float* Wv    = (const float*)d_in[3];
  const float* gamma = (const float*)d_in[4];
  float* out = (float*)d_out;

  // Workspace: Qh (1 MB) | Kh (1 MB) | Vt (8 MB), all f16.
  _Float16* Qh = (_Float16*)d_ws;
  _Float16* Kh = Qh + (size_t)CB * NN * CQK;
  _Float16* Vt = Kh + (size_t)CB * NN * CQK;

  qk_proj<<<CB * NN / 256, 256, 0, stream>>>(x, Wq, Wk, Qh, Kh);
  vt_proj<<<CB * NN / 64, 256, 0, stream>>>(x, Wv, Vt);
  attn_kernel<<<CB * NN / 16 / 8, 256, 0, stream>>>(x, Qh, Kh, Vt, gamma, out);
}
